// CausalSmokeModel_63015760167062
// MI455X (gfx1250) — matrix operand order repaired and checked by reading, not compiled, not run on hardware
//
#include <hip/hip_runtime.h>

// ---------------------------------------------------------------------------
// Model dims
// ---------------------------------------------------------------------------
#define BB   64
#define TT   512
#define DD   128
#define SS   16
#define PP   3
#define HH   512
#define KK   3
#define NN   510            // (T-P)/STRIDE + 1
#define TOK  (DD*PP)        // 384

typedef __attribute__((ext_vector_type(16))) __bf16 v16bf;
typedef __attribute__((ext_vector_type(8)))  __bf16 v8bf;
typedef __attribute__((ext_vector_type(8)))  float  v8f;

__device__ __forceinline__ unsigned short f2bf(float f) {
    unsigned int u = __builtin_bit_cast(unsigned int, f);
    u += 0x7fffu + ((u >> 16) & 1u);               // round-to-nearest-even
    return (unsigned short)(u >> 16);
}
__device__ __forceinline__ float bf2f(unsigned short h) {
    return __builtin_bit_cast(float, ((unsigned int)h) << 16);
}

// Build 16x32 bf16 A-fragment for one lane from row-major memory.
// lane L: row = base + (L&15); holds K = {kb..kb+15}, kb=(L>>4)*16
__device__ __forceinline__ v16bf load_a_frag(const __bf16* __restrict__ rowp, int kb) {
    v8bf lo = *(const v8bf*)(rowp + kb);
    v8bf hi = *(const v8bf*)(rowp + 8 + kb);
    return __builtin_shufflevector(lo, hi, 0,1,2,3,4,5,6,7,8,9,10,11,12,13,14,15);
}

#define WMMA_BF16(A_, B_, C_) \
    __builtin_amdgcn_wmma_f32_16x16x32_bf16(false, (A_), false, (B_), (short)0, (C_), false, false)

// ---------------------------------------------------------------------------
// f32 -> bf16 conversion
// ---------------------------------------------------------------------------
__global__ void cvt_f32_bf16(const float* __restrict__ in,
                             unsigned short* __restrict__ out, long long n) {
    long long i = (long long)blockIdx.x * blockDim.x + threadIdx.x;
    long long stride = (long long)gridDim.x * blockDim.x;
    for (; i < n; i += stride) out[i] = f2bf(in[i]);
}

// ---------------------------------------------------------------------------
// tokens gather: tokens[b,n,d*3+p] = aligned[b,n+p,d]; write fp32 out + bf16 copy
// ---------------------------------------------------------------------------
__global__ void tokens_gather(const float* __restrict__ aligned,
                              float* __restrict__ tok_f32,
                              unsigned short* __restrict__ tok_bf, long long total) {
    long long i = (long long)blockIdx.x * blockDim.x + threadIdx.x;
    long long stride = (long long)gridDim.x * blockDim.x;
    for (; i < total; i += stride) {
        long long b  = i / ((long long)NN * TOK);
        long long r  = i % ((long long)NN * TOK);
        long long n  = r / TOK;
        int kk = (int)(r % TOK);
        int d = kk / PP, p = kk % PP;
        float v = aligned[((size_t)b * TT + n + p) * DD + d];
        tok_f32[i] = v;
        tok_bf[i]  = f2bf(v);
    }
}

// ---------------------------------------------------------------------------
// Generic bf16 WMMA GEMM:  out[z][m, c] = A[z][m, :] . W[widx][c, :] + bias[c]
//   Per-wave register tile: 32 rows x 32 cols (4 wmma per k-step, 2 A + 2 B
//   fragment loads -> ~16 FLOP per cache byte).
//   grid.x = ceil(M/32), grid.y = batch z, grid.z = 512-col slab
//   block  = #cols in this slab (each wave owns 32 columns)
//   epi: 0 = fp32 store, 1 = RMS-norm over Nc then *rms_scale -> bf16 (single
//        slab only), 2 = exact GELU -> bf16
// ---------------------------------------------------------------------------
__global__ __launch_bounds__(512)
void gemm_bf16(const unsigned short* __restrict__ Abf,
               const unsigned short* __restrict__ Wbf,
               const float* __restrict__ bias,
               const float* __restrict__ rms_scale,
               void* __restrict__ out,
               int M, int Nc, int Kd, int lda, int ldo,
               long long A_zstride, long long W_zstride,
               long long bias_zstride, long long out_zstride,
               const int* __restrict__ wsel, int epi)
{
    __shared__ float ssum[32];

    int z    = blockIdx.y;
    int widx = wsel ? wsel[z] : z;
    const __bf16* A = (const __bf16*)Abf + (size_t)z * A_zstride;
    const __bf16* W = (const __bf16*)Wbf + (size_t)widx * W_zstride;
    const float* bz = bias + (size_t)widx * bias_zstride;
    size_t outz = (size_t)z * out_zstride;

    int wave = threadIdx.x >> 5, lane = threadIdx.x & 31;
    int mtile = blockIdx.x * 32;
    int am0 = mtile + (lane & 15);      if (am0 >= M) am0 = M - 1;   // clamp A rows
    int am1 = mtile + 16 + (lane & 15); if (am1 >= M) am1 = M - 1;
    int kb  = (lane >> 4) * 16;      // A k-subgroup
    int kb2 = (lane >> 4) * 16;      // B k-subgroup
    int hi8 = (lane >> 4) * 8;       // C/D row offset
    int c0  = blockIdx.z * 512 + wave * 32 + (lane & 15);            // out col, tile 0
    int c1  = c0 + 16;                                               // out col, tile 1

    v8f acc00 = {0.f,0.f,0.f,0.f,0.f,0.f,0.f,0.f};
    v8f acc01 = acc00, acc10 = acc00, acc11 = acc00;

    const __bf16* arow0 = A + (size_t)am0 * lda;
    const __bf16* arow1 = A + (size_t)am1 * lda;
    const __bf16* w0    = W + (size_t)c0 * Kd;
    const __bf16* w1    = W + (size_t)c1 * Kd;
    for (int k0 = 0; k0 < Kd; k0 += 32) {
        v16bf a0 = load_a_frag(arow0 + k0, kb);
        v16bf a1 = load_a_frag(arow1 + k0, kb);
        v16bf b0 = *(const v16bf*)(w0 + k0 + kb2);
        v16bf b1 = *(const v16bf*)(w1 + k0 + kb2);
        acc00 = WMMA_BF16(a0, b0, acc00);
        acc01 = WMMA_BF16(a0, b1, acc01);
        acc10 = WMMA_BF16(a1, b0, acc10);
        acc11 = WMMA_BF16(a1, b1, acc11);
    }

    float v00[8], v01[8], v10[8], v11[8];
    float bc0 = bz[c0], bc1 = bz[c1];
    #pragma unroll
    for (int r = 0; r < 8; ++r) {
        v00[r] = acc00[r] + bc0;  v01[r] = acc01[r] + bc1;
        v10[r] = acc10[r] + bc0;  v11[r] = acc11[r] + bc1;
    }

    if (epi == 1) {                       // RMS norm over the whole row (Nc cols)
        if (threadIdx.x < 32) ssum[threadIdx.x] = 0.f;
        __syncthreads();
        #pragma unroll
        for (int r = 0; r < 8; ++r) {
            atomicAdd(&ssum[hi8 + r],      v00[r] * v00[r] + v01[r] * v01[r]);
            atomicAdd(&ssum[16 + hi8 + r], v10[r] * v10[r] + v11[r] * v11[r]);
        }
        __syncthreads();
        float inv = 1.0f / (float)Nc;
        unsigned short* ob = (unsigned short*)out;
        float s0 = rms_scale[c0], s1 = rms_scale[c1];
        #pragma unroll
        for (int r = 0; r < 8; ++r) {
            int row0 = mtile + hi8 + r;
            int row1 = mtile + 16 + hi8 + r;
            if (row0 < M) {
                float ri = 1.0f / sqrtf(ssum[hi8 + r] * inv + 1e-6f);
                ob[outz + (size_t)row0 * ldo + c0] = f2bf(v00[r] * ri * s0);
                ob[outz + (size_t)row0 * ldo + c1] = f2bf(v01[r] * ri * s1);
            }
            if (row1 < M) {
                float ri = 1.0f / sqrtf(ssum[16 + hi8 + r] * inv + 1e-6f);
                ob[outz + (size_t)row1 * ldo + c0] = f2bf(v10[r] * ri * s0);
                ob[outz + (size_t)row1 * ldo + c1] = f2bf(v11[r] * ri * s1);
            }
        }
    } else if (epi == 2) {                // exact GELU -> bf16
        unsigned short* ob = (unsigned short*)out;
        const float is2 = 0.70710678118654752f;
        #pragma unroll
        for (int r = 0; r < 8; ++r) {
            int row0 = mtile + hi8 + r;
            int row1 = mtile + 16 + hi8 + r;
            if (row0 < M) {
                ob[outz + (size_t)row0 * ldo + c0] =
                    f2bf(0.5f * v00[r] * (1.0f + erff(v00[r] * is2)));
                ob[outz + (size_t)row0 * ldo + c1] =
                    f2bf(0.5f * v01[r] * (1.0f + erff(v01[r] * is2)));
            }
            if (row1 < M) {
                ob[outz + (size_t)row1 * ldo + c0] =
                    f2bf(0.5f * v10[r] * (1.0f + erff(v10[r] * is2)));
                ob[outz + (size_t)row1 * ldo + c1] =
                    f2bf(0.5f * v11[r] * (1.0f + erff(v11[r] * is2)));
            }
        }
    } else {                              // fp32 store
        float* ob = (float*)out;
        #pragma unroll
        for (int r = 0; r < 8; ++r) {
            int row0 = mtile + hi8 + r;
            int row1 = mtile + 16 + hi8 + r;
            if (row0 < M) {
                ob[outz + (size_t)row0 * ldo + c0] = v00[r];
                ob[outz + (size_t)row0 * ldo + c1] = v01[r];
            }
            if (row1 < M) {
                ob[outz + (size_t)row1 * ldo + c0] = v10[r];
                ob[outz + (size_t)row1 * ldo + c1] = v11[r];
            }
        }
    }
}

// ---------------------------------------------------------------------------
// GRU: one persistent workgroup (512 thr = 16 waves), 510 sequential steps.
// h state (64x512) lives in LDS as bf16.  Per step:
//   Phase A: gh = h_prev . Whh^T + bhh  (64x1536) via WMMA -> ghws (L2-resident)
//     wave = (mtPair, ntGroup): 2 row-tiles x 3 col-tiles per pass, 4 passes.
//     -> 6 wmma per (2 A + 3 B) fragment loads; every B fragment is loaded by
//        two waves per k-step (one per row block, wave&1 = 0 and 1).
//   Phase B: gates elementwise; h_new -> LDS + hidden_bf
// ---------------------------------------------------------------------------
__global__ __launch_bounds__(512)
void gru_kernel(const unsigned short* __restrict__ Whh_bf,
                const float* __restrict__ bhh,
                const float* __restrict__ gx,
                float* __restrict__ ghws,
                unsigned short* __restrict__ hidden_bf)
{
    __shared__ unsigned short hsh[BB * HH];      // 64 KB bf16 state

    int tid = threadIdx.x;
    int wave = tid >> 5, lane = tid & 31;
    for (int i = tid; i < BB * HH; i += 512) hsh[i] = 0;
    __syncthreads();

    const __bf16* Whh = (const __bf16*)Whh_bf;
    int kb  = (lane >> 4) * 16;
    int kb2 = (lane >> 4) * 16;
    int hi8 = (lane >> 4) * 8;
    int mrb = (wave & 1) * 32;             // row block base (2 tiles: mrb, mrb+16)
    int ntg = wave >> 1;                   // 0..7, owns cols [ntg*192, ntg*192+192)
    int m0  = mrb + (lane & 15);
    int m1  = m0 + 16;
    int bB    = tid >> 3;                  // batch row this thread owns in phase B
    int cbase = (tid & 7) * 64;            // 64 contiguous h-columns

    for (int n = 0; n < NN; ++n) {
        // ---- Phase A ----
        for (int p = 0; p < 4; ++p) {
            int cb  = ntg * 192 + p * 48;
            int wr0 = cb + (lane & 15);
            int wr1 = wr0 + 16;
            int wr2 = wr0 + 32;
            v8f acc00 = {0.f,0.f,0.f,0.f,0.f,0.f,0.f,0.f};
            v8f acc01 = acc00, acc02 = acc00;
            v8f acc10 = acc00, acc11 = acc00, acc12 = acc00;
            const __bf16* wp0 = Whh + (size_t)wr0 * HH;
            const __bf16* wp1 = Whh + (size_t)wr1 * HH;
            const __bf16* wp2 = Whh + (size_t)wr2 * HH;
            for (int k0 = 0; k0 < HH; k0 += 32) {
                v8bf lo0 = *(const v8bf*)(&hsh[m0 * HH + k0 + kb]);
                v8bf hi0 = *(const v8bf*)(&hsh[m0 * HH + k0 + 8 + kb]);
                v16bf a0 = __builtin_shufflevector(lo0, hi0,
                        0,1,2,3,4,5,6,7,8,9,10,11,12,13,14,15);
                v8bf lo1 = *(const v8bf*)(&hsh[m1 * HH + k0 + kb]);
                v8bf hi1 = *(const v8bf*)(&hsh[m1 * HH + k0 + 8 + kb]);
                v16bf a1 = __builtin_shufflevector(lo1, hi1,
                        0,1,2,3,4,5,6,7,8,9,10,11,12,13,14,15);
                v16bf b0 = *(const v16bf*)(wp0 + k0 + kb2);
                v16bf b1 = *(const v16bf*)(wp1 + k0 + kb2);
                v16bf b2 = *(const v16bf*)(wp2 + k0 + kb2);
                acc00 = WMMA_BF16(a0, b0, acc00);
                acc01 = WMMA_BF16(a0, b1, acc01);
                acc02 = WMMA_BF16(a0, b2, acc02);
                acc10 = WMMA_BF16(a1, b0, acc10);
                acc11 = WMMA_BF16(a1, b1, acc11);
                acc12 = WMMA_BF16(a1, b2, acc12);
            }
            float bc0 = bhh[wr0], bc1 = bhh[wr1], bc2 = bhh[wr2];
            #pragma unroll
            for (int r = 0; r < 8; ++r) {
                int row0 = mrb + hi8 + r;
                int row1 = mrb + 16 + hi8 + r;
                ghws[row0 * (3 * HH) + wr0] = acc00[r] + bc0;
                ghws[row0 * (3 * HH) + wr1] = acc01[r] + bc1;
                ghws[row0 * (3 * HH) + wr2] = acc02[r] + bc2;
                ghws[row1 * (3 * HH) + wr0] = acc10[r] + bc0;
                ghws[row1 * (3 * HH) + wr1] = acc11[r] + bc1;
                ghws[row1 * (3 * HH) + wr2] = acc12[r] + bc2;
            }
        }
        __threadfence_block();
        __syncthreads();

        // ---- Phase B: gates ----
        size_t gxb = ((size_t)(bB * NN + n)) * (3 * HH);
        size_t hob = ((size_t)(bB * NN + n)) * HH;
        int ghb = bB * (3 * HH);
        for (int j = 0; j < 64; ++j) {
            int c = cbase + j;
            float r_ = 1.f / (1.f + expf(-(gx[gxb + c]          + ghws[ghb + c])));
            float z_ = 1.f / (1.f + expf(-(gx[gxb + HH + c]     + ghws[ghb + HH + c])));
            float n_ = tanhf(gx[gxb + 2 * HH + c] + r_ * ghws[ghb + 2 * HH + c]);
            float hp = bf2f(hsh[bB * HH + c]);
            float hn = (1.f - z_) * n_ + z_ * hp;
            unsigned short hb = f2bf(hn);
            hsh[bB * HH + c]   = hb;
            hidden_bf[hob + c] = hb;
        }
        __threadfence_block();
        __syncthreads();
    }
}

// ---------------------------------------------------------------------------
// host launcher
// ---------------------------------------------------------------------------
extern "C" void kernel_launch(void* const* d_in, const int* in_sizes, int n_in,
                              void* d_out, int out_size, void* d_ws, size_t ws_size,
                              hipStream_t stream) {
    const float* x         = (const float*)d_in[0];
    const int*   sidx      = (const int*)  d_in[1];
    const float* sess_W    = (const float*)d_in[2];
    const float* sess_b    = (const float*)d_in[3];
    const float* proj_W    = (const float*)d_in[4];
    const float* proj_b    = (const float*)d_in[5];
    const float* rms_scale = (const float*)d_in[6];
    const float* gru_Wih   = (const float*)d_in[7];
    const float* gru_Whh   = (const float*)d_in[8];
    const float* gru_bih   = (const float*)d_in[9];
    const float* gru_bhh   = (const float*)d_in[10];
    const float* head_W1   = (const float*)d_in[11];
    const float* head_b1   = (const float*)d_in[12];
    const float* head_W2   = (const float*)d_in[13];
    const float* head_b2   = (const float*)d_in[14];

    char* base = (char*)d_ws;
    size_t off = 0;
    auto alloc = [&](size_t bytes) -> char* {
        char* p = base + off;
        off = (off + bytes + 255) & ~(size_t)255;
        return p;
    };

    unsigned short* x_bf     = (unsigned short*)alloc((size_t)BB*TT*DD*2);
    unsigned short* sessW_bf = (unsigned short*)alloc((size_t)SS*DD*DD*2);
    unsigned short* projW_bf = (unsigned short*)alloc((size_t)HH*TOK*2);
    unsigned short* Wih_bf   = (unsigned short*)alloc((size_t)3*HH*HH*2);
    unsigned short* Whh_bf   = (unsigned short*)alloc((size_t)3*HH*HH*2);
    unsigned short* W1_bf    = (unsigned short*)alloc((size_t)KK*HH*HH*2);
    unsigned short* W2_bf    = (unsigned short*)alloc((size_t)KK*TOK*HH*2);
    float*          aligned  = (float*)        alloc((size_t)BB*TT*DD*4);
    unsigned short* tok_bf   = (unsigned short*)alloc((size_t)BB*NN*TOK*2);
    unsigned short* h_bf     = (unsigned short*)alloc((size_t)BB*NN*HH*2);
    char*           gx_raw   =                 alloc((size_t)BB*NN*3*HH*4);
    unsigned short* hid_bf   = (unsigned short*)alloc((size_t)BB*NN*HH*2);
    float*          ghws     = (float*)        alloc((size_t)BB*3*HH*4);
    float*          gx       = (float*)gx_raw;
    unsigned short* a_bf     = (unsigned short*)gx_raw;  // alias: gx dead after GRU

    // 1) bf16 conversions
    auto cvt = [&](const float* src, unsigned short* dst, long long n) {
        int grid = (int)((n + 255) / 256); if (grid > 4096) grid = 4096;
        cvt_f32_bf16<<<grid, 256, 0, stream>>>(src, dst, n);
    };
    cvt(x,        x_bf,     (long long)BB*TT*DD);
    cvt(sess_W,   sessW_bf, (long long)SS*DD*DD);
    cvt(proj_W,   projW_bf, (long long)HH*TOK);
    cvt(gru_Wih,  Wih_bf,   (long long)3*HH*HH);
    cvt(gru_Whh,  Whh_bf,   (long long)3*HH*HH);
    cvt(head_W1,  W1_bf,    (long long)KK*HH*HH);
    cvt(head_W2,  W2_bf,    (long long)KK*TOK*HH);

    // 2) session align: aligned[b] = x[b] . sess_W[sidx[b]]^T + sess_b[sidx[b]]
    gemm_bf16<<<dim3(TT/32, BB, 1), 128, 0, stream>>>(
        x_bf, sessW_bf, sess_b, nullptr, aligned,
        TT, DD, DD, DD, DD,
        (long long)TT*DD, (long long)DD*DD, (long long)DD, (long long)TT*DD,
        sidx, 0);

    // 3) tokens gather (fp32 to d_out tail + bf16 copy)
    {
        long long total = (long long)BB*NN*TOK;
        float* tok_out = (float*)d_out + (size_t)KK*BB*NN*TOK;
        int grid = (int)((total + 255) / 256);
        tokens_gather<<<grid, 256, 0, stream>>>(aligned, tok_out, tok_bf, total);
    }

    // 4) h = RMSnorm(tokens . proj_W^T + proj_b) * rms_scale   -> bf16
    gemm_bf16<<<dim3((NN+31)/32, BB, 1), 512, 0, stream>>>(
        tok_bf, projW_bf, proj_b, rms_scale, h_bf,
        NN, HH, TOK, TOK, HH,
        (long long)NN*TOK, 0, 0, (long long)NN*HH,
        nullptr, 1);

    // 5) gx = h . Wih^T + bih  (fp32, 3 column slabs)
    gemm_bf16<<<dim3((NN+31)/32, BB, 3), 512, 0, stream>>>(
        h_bf, Wih_bf, gru_bih, nullptr, gx,
        NN, 3*HH, HH, HH, 3*HH,
        (long long)NN*HH, 0, 0, (long long)NN*3*HH,
        nullptr, 0);

    // 6) GRU recurrence (single persistent workgroup)
    gru_kernel<<<1, 512, 0, stream>>>(Whh_bf, gru_bhh, gx, ghws, hid_bf);

    // 7) heads, per horizon k
    for (int k = 0; k < KK; ++k) {
        // a = gelu(hidden . W1[k]^T + b1[k]) -> bf16 (aliases dead gx space)
        gemm_bf16<<<dim3((NN+31)/32, BB, 1), 512, 0, stream>>>(
            hid_bf, W1_bf + (size_t)k*HH*HH, head_b1 + (size_t)k*HH, nullptr,
            a_bf + (size_t)k*BB*NN*HH,
            NN, HH, HH, HH, HH,
            (long long)NN*HH, 0, 0, (long long)NN*HH,
            nullptr, 2);
        // preds[k] = a . W2[k]^T + b2[k] -> fp32 into d_out
        gemm_bf16<<<dim3((NN+31)/32, BB, 1), 384, 0, stream>>>(
            a_bf + (size_t)k*BB*NN*HH, W2_bf + (size_t)k*TOK*HH,
            head_b2 + (size_t)k*TOK, nullptr,
            (float*)d_out + (size_t)k*BB*NN*TOK,
            NN, TOK, HH, HH, TOK,
            (long long)NN*HH, 0, 0, (long long)NN*TOK,
            nullptr, 0);
    }
}
